// Self_attention_82291573391881
// MI455X (gfx1250) — matrix operand order repaired and checked by reading, not compiled, not run on hardware
//
#include <hip/hip_runtime.h>

typedef __attribute__((ext_vector_type(16))) _Float16 v16h;
typedef __attribute__((ext_vector_type(8)))  _Float16 v8h;
typedef __attribute__((ext_vector_type(8)))  float    v8f;
typedef __attribute__((ext_vector_type(4)))  float    f32x4;

#define WMMA(a,b,c) __builtin_amdgcn_wmma_f32_16x16x32_f16(false,(a),false,(b),(short)0,(c),false,false)

namespace {
constexpr int S = 2048;
constexpr int D = 64;
constexpr int HEADS = 16;
constexpr int BATCH = 4;
constexpr int WAVES = 8;          // waves per block
constexpr int PSTRIDE = 40;       // f16 per LDS row (80B => 16B-aligned rows, odd bank phase)
constexpr float SCALE   = 0.125f; // 1/sqrt(64)
constexpr float NEGINIT = -1e30f;
constexpr float MASKNEG = -1000000000.0f;
}

__device__ __forceinline__ v16h ld16f(const float* p) {
  // 16 contiguous floats -> 16 f16 halves (half h == element h)
  const f32x4* p4 = (const f32x4*)p;
  f32x4 a = p4[0], b = p4[1], c = p4[2], d = p4[3];
  v16h r;
#pragma unroll
  for (int i = 0; i < 4; ++i) {
    r[i]      = (_Float16)a[i];
    r[4 + i]  = (_Float16)b[i];
    r[8 + i]  = (_Float16)c[i];
    r[12 + i] = (_Float16)d[i];
  }
  return r;
}

__global__ __launch_bounds__(256) void attn_fa_kernel(
    const float* __restrict__ Kp, const float* __restrict__ Qp,
    const float* __restrict__ Vp, const int* __restrict__ Mp,
    float* __restrict__ Op)
{
  __shared__ _Float16 pbuf[WAVES][16 * PSTRIDE];

  const int lane = threadIdx.x & 31;
  const int wave = threadIdx.x >> 5;
  const int gw   = blockIdx.x * WAVES + wave;     // 0..8191
  const int qtile = gw & (S / 16 - 1);            // 128 q-tiles per (b,h)
  const int bh    = gw >> 7;
  const int b     = bh >> 4;                      // 16 heads
  const int qbase = qtile * 16;
  const int n  = lane & 15;
  const int hi = lane >> 4;

  // ---- persistent Q tile: two 16x32 f16 A-fragments ----
  // A layout: lane holds row M=lane&15; halves h -> k = 16*hi + h
  const float* qrow = Qp + ((size_t)bh * S + qbase + n) * D;
  const f32x4* q4 = (const f32x4*)qrow;
  v16h qa0, qa1;
  {
    f32x4 r0 = q4[4*hi + 0], r1 = q4[4*hi + 1];
    f32x4 r2 = q4[4*hi + 2], r3 = q4[4*hi + 3];
    f32x4 r4 = q4[4*hi + 8], r5 = q4[4*hi + 9];
    f32x4 r6 = q4[4*hi +10], r7 = q4[4*hi +11];
#pragma unroll
    for (int i = 0; i < 4; ++i) {
      qa0[i]      = (_Float16)r0[i];
      qa0[4 + i]  = (_Float16)r1[i];
      qa0[8 + i]  = (_Float16)r2[i];
      qa0[12 + i] = (_Float16)r3[i];
      qa1[i]      = (_Float16)r4[i];
      qa1[4 + i]  = (_Float16)r5[i];
      qa1[8 + i]  = (_Float16)r6[i];
      qa1[12 + i] = (_Float16)r7[i];
    }
  }

  v8f acc0 = {}, acc1 = {}, acc2 = {}, acc3 = {};
  float mrun[8], lrun[8];
#pragma unroll
  for (int v = 0; v < 8; ++v) { mrun[v] = NEGINIT; lrun[v] = 0.0f; }

  const float* kbase = Kp + (size_t)bh * S * D;
  const float* vbase = Vp + (size_t)bh * S * D;
  // C-layout rows: lanes 0-15 hold rows v (0..7), lanes 16-31 rows v+8
  const int* mbase = Mp + (size_t)b * S * S + (size_t)(qbase + hi * 8) * S + n;

  for (int j0 = 0; j0 < S; j0 += 32) {
    // ---- K tile as B-fragments: B[d][key]; lane = key + 16*(d_local>>4), half = d_local&15
    const float* krow0 = kbase + (size_t)(j0 + n) * D;        // keys j0..j0+15
    const float* krow1 = krow0 + 16 * D;                      // keys j0+16..j0+31
    v16h kb0l = ld16f(krow0 + hi * 16);        // d 0..31
    v16h kb0h = ld16f(krow0 + 32 + hi * 16);   // d 32..63
    v16h kb1l = ld16f(krow1 + hi * 16);
    v16h kb1h = ld16f(krow1 + 32 + hi * 16);

    v8f z = {};
    v8f s0 = WMMA(qa1, kb0h, WMMA(qa0, kb0l, z));   // scores for keys j0..j0+15
    v8f s1 = WMMA(qa1, kb1h, WMMA(qa0, kb1l, z));   // scores for keys j0+16..j0+31

    // ---- scale + mask (mask is head-independent) ----
    const int* mp = mbase + j0;
#pragma unroll
    for (int v = 0; v < 8; ++v) {
      int m0 = mp[(size_t)v * S];
      int m1 = mp[(size_t)v * S + 16];
      float a0 = s0[v] * SCALE; if (m0 == 0) a0 = MASKNEG;
      float a1 = s1[v] * SCALE; if (m1 == 0) a1 = MASKNEG;
      s0[v] = a0; s1[v] = a1;
    }

    // ---- online softmax (row reductions across 16-lane half-groups) ----
#pragma unroll
    for (int v = 0; v < 8; ++v) {
      float t = fmaxf(s0[v], s1[v]);
#pragma unroll
      for (int off = 1; off < 16; off <<= 1)
        t = fmaxf(t, __shfl_xor(t, off, 32));
      float mn = fmaxf(mrun[v], t);
      float alpha = __expf(mrun[v] - mn);
      mrun[v] = mn;
      float p0 = __expf(s0[v] - mn);
      float p1 = __expf(s1[v] - mn);
      float rs = p0 + p1;
#pragma unroll
      for (int off = 1; off < 16; off <<= 1)
        rs += __shfl_xor(rs, off, 32);
      lrun[v] = lrun[v] * alpha + rs;
      acc0[v] *= alpha; acc1[v] *= alpha; acc2[v] *= alpha; acc3[v] *= alpha;
      s0[v] = p0; s1[v] = p1;
    }

    // ---- C-layout P -> A-layout P via LDS (f16) ----
    _Float16* pb = &pbuf[wave][0];
    asm volatile("" ::: "memory");
#pragma unroll
    for (int v = 0; v < 8; ++v) {
      int row = v + 8 * hi;
      pb[row * PSTRIDE + n]      = (_Float16)s0[v];
      pb[row * PSTRIDE + 16 + n] = (_Float16)s1[v];
    }
    asm volatile("s_wait_dscnt 0" ::: "memory");   // DS ops are in-order per wave
    v16h pa;
    {
      const _Float16* pr = pb + n * PSTRIDE + hi * 16;   // 16B-aligned (PSTRIDE*2 = 80 = 16*5)
      v8h lo  = *(const v8h*)pr;          // k = 16*hi .. 16*hi+7
      v8h hi8 = *(const v8h*)(pr + 8);    // k = 16*hi+8 .. 16*hi+15
#pragma unroll
      for (int i = 0; i < 8; ++i) { pa[i] = lo[i]; pa[8 + i] = hi8[i]; }
    }
    asm volatile("" ::: "memory");

    // ---- V tile as B-fragments: lane holds k_local = hi*16+h, d = c*16+n ----
    const float* vp = vbase + (size_t)(j0 + hi * 16) * D + n;
    if (j0 + 32 < S) {                        // prefetch next K/V tiles
      __builtin_prefetch(krow0 + 32 * D, 0, 0);
      __builtin_prefetch(vp + 32 * D, 0, 0);
    }
    v16h vb0, vb1, vb2, vb3;
#pragma unroll
    for (int h = 0; h < 16; ++h) {
      const float* r = vp + (size_t)h * D;
      vb0[h] = (_Float16)r[0];
      vb1[h] = (_Float16)r[16];
      vb2[h] = (_Float16)r[32];
      vb3[h] = (_Float16)r[48];
    }

    acc0 = WMMA(pa, vb0, acc0);
    acc1 = WMMA(pa, vb1, acc1);
    acc2 = WMMA(pa, vb2, acc2);
    acc3 = WMMA(pa, vb3, acc3);
  }

  // ---- epilogue: normalize by running sum, coalesced fp32 stores ----
  float* orow = Op + ((size_t)bh * S + qbase + hi * 8) * D + n;
#pragma unroll
  for (int v = 0; v < 8; ++v) {
    float inv = 1.0f / lrun[v];
    float* o = orow + (size_t)v * D;
    o[0]  = acc0[v] * inv;
    o[16] = acc1[v] * inv;
    o[32] = acc2[v] * inv;
    o[48] = acc3[v] * inv;
  }
}

extern "C" void kernel_launch(void* const* d_in, const int* in_sizes, int n_in,
                              void* d_out, int out_size, void* d_ws, size_t ws_size,
                              hipStream_t stream) {
  // setup_inputs order: key, query, value, mask
  const float* Kp = (const float*)d_in[0];
  const float* Qp = (const float*)d_in[1];
  const float* Vp = (const float*)d_in[2];
  const int*   Mp = (const int*)d_in[3];
  float* Op = (float*)d_out;

  const int total_waves = BATCH * HEADS * (S / 16);   // 8192
  dim3 grid(total_waves / WAVES);                     // 1024 blocks
  dim3 block(WAVES * 32);                             // 256 threads
  attn_fa_kernel<<<grid, block, 0, stream>>>(Kp, Qp, Vp, Mp, Op);
}
